// Seq2SeqLSTMModel_20160576487519
// MI455X (gfx1250) — compile-verified, matrix operand order checked by reading, not run on hardware
//
#include <hip/hip_runtime.h>
#include <hip/hip_bf16.h>
#include <cstdint>

// ---------------------------------------------------------------------------
// Seq2Seq LSTM (enc 2x64, 50 steps; dec 2x64, 60 steps; linear 64->2)
// CDNA5 / gfx1250: wave32 WMMA f32_16x16x32_f16, weights staged in LDS in
// B-fragment layout, cell state kept in VGPRs in D-fragment layout.
// ---------------------------------------------------------------------------

typedef __attribute__((ext_vector_type(16))) _Float16 v16h;
typedef __attribute__((ext_vector_type(8)))  float    v8f;

#define B_TOT    32768
#define N_AGENT  10
#define SEQ_LEN  50
#define N_IN     6
#define HID      64
#define N_OUT    2
#define T_DEC    60

#define WAVES_PER_WG   8
#define ROWS_PER_WAVE  16
#define ROWS_PER_WG    (WAVES_PER_WG * ROWS_PER_WAVE)   // 128 -> 256 workgroups

#define HSTRIDE 66   // padded _Float16 row stride (33 dwords, odd -> conflict-free)

// Weight fragment store: [kTile][nTile(16)][v(8)][lane(32)] packed f16 pairs
#define WFRAG_L0_DW (3 * 16 * 8 * 32)   // 12288 dwords (x-pad32 + h 2x32)
#define WFRAG_L1_DW (4 * 16 * 8 * 32)   // 16384 dwords (h0 2x32 + h1 2x32)
#define WFRAG_DW    (WFRAG_L0_DW + WFRAG_L1_DW)

#define SMEM_BYTES (WFRAG_DW * 4                 /* weight fragments 114688 */ \
                  + 4 * 256 * 4                  /* 4 combined biases 4096  */ \
                  + 128 * 4 + 4 * 4              /* lin_W + lin_b           */ \
                  + WAVES_PER_WG * 32 * 4        /* decoder feedback x      */ \
                  + WAVES_PER_WG * 2 * ROWS_PER_WAVE * HSTRIDE * 2 /* h0/h1 */)

union FragH {
  v16h      v;
  uint32_t  u[8];
  _Float16  h[16];
};

// ISA 7.12.2 16-bit A/B layout: lane half (0/1) and reg v select K offset
__device__ __forceinline__ int koff_of(int v, int half) {
  return ((v & 4) << 2) + (half << 3) + ((v & 3) << 1);   // {0,16} + 8*half + 2*(v%4)
}

__device__ __forceinline__ float sigm(float x) { return 1.0f / (1.0f + __expf(-x)); }

// --- fragment builders ------------------------------------------------------

// B fragment: 8 conflict-free ds_load_b32 (lane-major layout)
__device__ __forceinline__ void load_bfrag(FragH& f, const uint32_t* wf, int kt, int nt, int lane) {
  const uint32_t* p = wf + ((kt * 16 + nt) * 8) * 32 + lane;
#pragma unroll
  for (int v = 0; v < 8; ++v) f.u[v] = p[v * 32];
}

// A fragment from an LDS h-buffer (16 rows x 64 f16, row stride HSTRIDE)
__device__ __forceinline__ void build_afrag_h(FragH& f, const _Float16* hbuf, int ktLocal, int lane) {
  const int m = lane & 15, half = lane >> 4;
  const uint32_t* hb = (const uint32_t*)hbuf;
#pragma unroll
  for (int v = 0; v < 8; ++v) {
    int k = koff_of(v, half) + ktLocal * 32;           // always even
    f.u[v] = hb[m * (HSTRIDE / 2) + (k >> 1)];
  }
}

// Encoder input fragment straight from global history (K padded 6 -> 32)
__device__ __forceinline__ void build_afrag_xenc(FragH& f, const float* hist, int rowBase, int t, int lane) {
  const int m = lane & 15, half = lane >> 4;
  const float* p = hist + (size_t)(rowBase + m) * (N_AGENT * SEQ_LEN * N_IN) + (size_t)t * N_IN;
#pragma unroll
  for (int v = 0; v < 8; ++v) {
    int k = koff_of(v, half);
    f.h[2 * v]     = (k     < N_IN) ? (_Float16)p[k]     : (_Float16)0.0f;
    f.h[2 * v + 1] = (k + 1 < N_IN) ? (_Float16)p[k + 1] : (_Float16)0.0f;
  }
}

// Decoder input fragment from LDS feedback buffer (K padded 2 -> 32)
__device__ __forceinline__ void build_afrag_xdec(FragH& f, const float* xdec, int lane) {
  const int m = lane & 15, half = lane >> 4;
#pragma unroll
  for (int v = 0; v < 8; ++v) {
    int k = koff_of(v, half);
    f.h[2 * v]     = (k     < N_OUT) ? (_Float16)xdec[m * 2 + k]     : (_Float16)0.0f;
    f.h[2 * v + 1] = (k + 1 < N_OUT) ? (_Float16)xdec[m * 2 + k + 1] : (_Float16)0.0f;
  }
}

// --- one LSTM layer step: gates -> (c,h) update ----------------------------
// KT = number of 32-wide K tiles (3 for layer0, 4 for layer1). A fragments are
// hoisted so hOut can be overwritten safely. c state stays in VGPRs (D layout).
template <int KT>
__device__ __forceinline__ void layer_step(const uint32_t* wf, const float* bias,
                                           const FragH (&a)[KT], float (&cst)[4][8],
                                           _Float16* hOut, int lane) {
  const int n = lane & 15, half = lane >> 4;
#pragma unroll
  for (int j = 0; j < 4; ++j) {
    const float bi = bias[      j * 16 + n];
    const float bf = bias[ 64 + j * 16 + n];
    const float bg = bias[128 + j * 16 + n];
    const float bo = bias[192 + j * 16 + n];
    v8f ai, af, ag, ao;
#pragma unroll
    for (int v = 0; v < 8; ++v) { ai[v] = bi; af[v] = bf; ag[v] = bg; ao[v] = bo; }
#pragma unroll
    for (int kt = 0; kt < KT; ++kt) {
      FragH bw;
      load_bfrag(bw, wf, kt,      j, lane);
      ai = __builtin_amdgcn_wmma_f32_16x16x32_f16(false, a[kt].v, false, bw.v, (short)0, ai, false, false);
      load_bfrag(bw, wf, kt,  4 + j, lane);
      af = __builtin_amdgcn_wmma_f32_16x16x32_f16(false, a[kt].v, false, bw.v, (short)0, af, false, false);
      load_bfrag(bw, wf, kt,  8 + j, lane);
      ag = __builtin_amdgcn_wmma_f32_16x16x32_f16(false, a[kt].v, false, bw.v, (short)0, ag, false, false);
      load_bfrag(bw, wf, kt, 12 + j, lane);
      ao = __builtin_amdgcn_wmma_f32_16x16x32_f16(false, a[kt].v, false, bw.v, (short)0, ao, false, false);
    }
#pragma unroll
    for (int v = 0; v < 8; ++v) {
      float iv = sigm(ai[v]);
      float fv = sigm(af[v]);
      float gv = tanhf(ag[v]);
      float ov = sigm(ao[v]);
      float cn = fv * cst[j][v] + iv * gv;
      cst[j][v] = cn;
      float hv = ov * tanhf(cn);
      int   m  = half * 8 + v;                       // D layout row
      hOut[m * HSTRIDE + j * 16 + n] = (_Float16)hv;
    }
  }
}

// --- weight staging: global f32 -> LDS f16 in B-fragment layout -------------
__device__ void stage_weights(uint32_t* wfrag,
                              const float* Wih0, const float* Whh0,
                              const float* Wih1, const float* Whh1, int K0real) {
  const int tid = threadIdx.x;
  // layer0: kt0 = Wih0 (K padded to 32), kt1..2 = Whh0
  for (int w = tid; w < WFRAG_L0_DW; w += blockDim.x) {
    int lane = w & 31, v = (w >> 5) & 7, nt = (w >> 8) & 15, kt = w >> 12;
    int n = nt * 16 + (lane & 15), half = lane >> 4;
    int koff = koff_of(v, half);
    _Float16 lo, hi;
    if (kt == 0) {
      lo = (koff     < K0real) ? (_Float16)Wih0[n * K0real + koff]     : (_Float16)0.0f;
      hi = (koff + 1 < K0real) ? (_Float16)Wih0[n * K0real + koff + 1] : (_Float16)0.0f;
    } else {
      int k = (kt - 1) * 32 + koff;
      lo = (_Float16)Whh0[n * HID + k];
      hi = (_Float16)Whh0[n * HID + k + 1];
    }
    union { uint32_t u; _Float16 h[2]; } pk;
    pk.h[0] = lo; pk.h[1] = hi;
    wfrag[w] = pk.u;
  }
  // layer1: kt0..1 = Wih1, kt2..3 = Whh1 (both K=64)
  for (int w = tid; w < WFRAG_L1_DW; w += blockDim.x) {
    int lane = w & 31, v = (w >> 5) & 7, nt = (w >> 8) & 15, kt = w >> 12;
    int n = nt * 16 + (lane & 15), half = lane >> 4;
    int k = (kt & 1) * 32 + koff_of(v, half);
    const float* W = (kt < 2) ? Wih1 : Whh1;
    union { uint32_t u; _Float16 h[2]; } pk;
    pk.h[0] = (_Float16)W[n * HID + k];
    pk.h[1] = (_Float16)W[n * HID + k + 1];
    wfrag[WFRAG_L0_DW + w] = pk.u;
  }
}

// ---------------------------------------------------------------------------
__global__ void __launch_bounds__(256, 1)
seq2seq_lstm_kernel(const float* __restrict__ hist,
                    const float* eWih0, const float* eWhh0, const float* ebih0, const float* ebhh0,
                    const float* eWih1, const float* eWhh1, const float* ebih1, const float* ebhh1,
                    const float* dWih0, const float* dWhh0, const float* dbih0, const float* dbhh0,
                    const float* dWih1, const float* dWhh1, const float* dbih1, const float* dbhh1,
                    const float* linW, const float* linb, const float* start,
                    float* __restrict__ out) {
  extern __shared__ char smem[];
  uint32_t* wfrag  = (uint32_t*)smem;
  float*    biasE0 = (float*)(wfrag + WFRAG_DW);
  float*    biasE1 = biasE0 + 256;
  float*    biasD0 = biasE1 + 256;
  float*    biasD1 = biasD0 + 256;
  float*    linWs  = biasD1 + 256;          // 128 floats
  float*    linbs  = linWs + 128;           // 4 floats (2 used)
  float*    xdecAll = linbs + 4;            // 8 waves * 32 floats
  _Float16* hAll   = (_Float16*)(xdecAll + WAVES_PER_WG * 32);

  const int tid  = threadIdx.x;
  const int lane = tid & 31;
  const int wave = tid >> 5;
  const int rowBase = blockIdx.x * ROWS_PER_WG + wave * ROWS_PER_WAVE;

  _Float16* h0buf = hAll + wave * (2 * ROWS_PER_WAVE * HSTRIDE);
  _Float16* h1buf = h0buf + ROWS_PER_WAVE * HSTRIDE;
  float*    xdec  = xdecAll + wave * 32;

  // ---- stage encoder weights, all biases, linear weights; zero h ----------
  stage_weights(wfrag, eWih0, eWhh0, eWih1, eWhh1, N_IN);
  for (int n = tid; n < 256; n += blockDim.x) {
    biasE0[n] = ebih0[n] + ebhh0[n];
    biasE1[n] = ebih1[n] + ebhh1[n];
    biasD0[n] = dbih0[n] + dbhh0[n];
    biasD1[n] = dbih1[n] + dbhh1[n];
  }
  for (int n = tid; n < 128; n += blockDim.x) linWs[n] = linW[n];
  if (tid < 2) linbs[tid] = linb[tid];
  for (int n = tid; n < WAVES_PER_WG * 2 * ROWS_PER_WAVE * HSTRIDE; n += blockDim.x)
    hAll[n] = (_Float16)0.0f;
  __syncthreads();

  float c0[4][8] = {};   // layer-0 cell state, D-fragment layout, lives in VGPRs
  float c1[4][8] = {};   // layer-1 cell state

  // ---- encoder: 50 steps ---------------------------------------------------
  for (int t = 0; t < SEQ_LEN; ++t) {
    FragH a0[3];
    build_afrag_xenc(a0[0], hist, rowBase, t, lane);
    build_afrag_h(a0[1], h0buf, 0, lane);
    build_afrag_h(a0[2], h0buf, 1, lane);
    layer_step<3>(wfrag, biasE0, a0, c0, h0buf, lane);
    __syncthreads();

    FragH a1[4];
    build_afrag_h(a1[0], h0buf, 0, lane);
    build_afrag_h(a1[1], h0buf, 1, lane);
    build_afrag_h(a1[2], h1buf, 0, lane);
    build_afrag_h(a1[3], h1buf, 1, lane);
    layer_step<4>(wfrag + WFRAG_L0_DW, biasE1, a1, c1, h1buf, lane);
    __syncthreads();
  }

  // ---- restage with decoder weights; seed feedback with start token -------
  stage_weights(wfrag, dWih0, dWhh0, dWih1, dWhh1, N_OUT);
  {
    int r = lane >> 1, o = lane & 1;
    xdec[r * 2 + o] = start[o];
  }
  __syncthreads();

  // ---- decoder: 60 steps, feed output back as next input -------------------
  for (int t = 0; t < T_DEC; ++t) {
    FragH a0[3];
    build_afrag_xdec(a0[0], xdec, lane);
    build_afrag_h(a0[1], h0buf, 0, lane);
    build_afrag_h(a0[2], h0buf, 1, lane);
    layer_step<3>(wfrag, biasD0, a0, c0, h0buf, lane);
    __syncthreads();

    FragH a1[4];
    build_afrag_h(a1[0], h0buf, 0, lane);
    build_afrag_h(a1[1], h0buf, 1, lane);
    build_afrag_h(a1[2], h1buf, 0, lane);
    build_afrag_h(a1[3], h1buf, 1, lane);
    layer_step<4>(wfrag + WFRAG_L0_DW, biasD1, a1, c1, h1buf, lane);
    __syncthreads();

    // linear 64->2: one output scalar per lane (r = lane/2, o = lane%2)
    {
      int r = lane >> 1, o = lane & 1;
      float acc = linbs[o];
#pragma unroll
      for (int k = 0; k < HID; ++k)
        acc += (float)h1buf[r * HSTRIDE + k] * linWs[o * HID + k];
      xdec[r * 2 + o] = acc;
      out[((size_t)(rowBase + r) * T_DEC + t) * N_OUT + o] = acc;
    }
    __syncthreads();
  }
}

// ---------------------------------------------------------------------------
extern "C" void kernel_launch(void* const* d_in, const int* in_sizes, int n_in,
                              void* d_out, int out_size, void* d_ws, size_t ws_size,
                              hipStream_t stream) {
  (void)in_sizes; (void)n_in; (void)out_size; (void)d_ws; (void)ws_size;
  const float* hist  = (const float*)d_in[0];
  const float* eWih0 = (const float*)d_in[1];
  const float* eWhh0 = (const float*)d_in[2];
  const float* ebih0 = (const float*)d_in[3];
  const float* ebhh0 = (const float*)d_in[4];
  const float* eWih1 = (const float*)d_in[5];
  const float* eWhh1 = (const float*)d_in[6];
  const float* ebih1 = (const float*)d_in[7];
  const float* ebhh1 = (const float*)d_in[8];
  const float* dWih0 = (const float*)d_in[9];
  const float* dWhh0 = (const float*)d_in[10];
  const float* dbih0 = (const float*)d_in[11];
  const float* dbhh0 = (const float*)d_in[12];
  const float* dWih1 = (const float*)d_in[13];
  const float* dWhh1 = (const float*)d_in[14];
  const float* dbih1 = (const float*)d_in[15];
  const float* dbhh1 = (const float*)d_in[16];
  const float* linW  = (const float*)d_in[17];
  const float* linb  = (const float*)d_in[18];
  const float* start = (const float*)d_in[19];

  dim3 grid(B_TOT / ROWS_PER_WG);          // 256 workgroups
  dim3 block(WAVES_PER_WG * 32);           // 256 threads = 8 waves
  hipLaunchKernelGGL(seq2seq_lstm_kernel, grid, block, SMEM_BYTES, stream,
                     hist, eWih0, eWhh0, ebih0, ebhh0, eWih1, eWhh1, ebih1, ebhh1,
                     dWih0, dWhh0, dbih0, dbhh0, dWih1, dWhh1, dbih1, dbhh1,
                     linW, linb, start, (float*)d_out);
}
